// SelfAttention_43868795961965
// MI455X (gfx1250) — matrix operand order repaired and checked by reading, not compiled, not run on hardware
//
#include <hip/hip_runtime.h>
#include <hip/hip_bf16.h>

typedef __attribute__((ext_vector_type(16))) __bf16 v16bf;
typedef __attribute__((ext_vector_type(8)))  __bf16 v8bf;
typedef __attribute__((ext_vector_type(8)))  float  v8f;

#define EPS_SN 1e-12f

// ---------------------------------------------------------------------------
// sigma = || w @ normalize(w^T u) ||  (one power iteration, v-vector unused)
// ---------------------------------------------------------------------------
__global__ void sn_sigma_kernel(const float* __restrict__ w,
                                const float* __restrict__ u,
                                int D, int C,
                                float* __restrict__ inv_sigma) {
    __shared__ float sv[256];
    __shared__ float red[256];
    const int tid = threadIdx.x;

    float t = 0.f;
    if (tid < C) {
        for (int d = 0; d < D; ++d) t = fmaf(w[d * C + tid], u[d], t);
    }
    red[tid] = t * t;
    __syncthreads();
    for (int s = 128; s > 0; s >>= 1) {
        if (tid < s) red[tid] += red[tid + s];
        __syncthreads();
    }
    const float tn = sqrtf(red[0]) + EPS_SN;
    sv[tid] = (tid < C) ? (t / tn) : 0.f;
    __syncthreads();

    float y = 0.f;
    if (tid < D) {
        for (int c = 0; c < C; ++c) y = fmaf(w[tid * C + c], sv[c], y);
    }
    red[tid] = y * y;
    __syncthreads();
    for (int s = 128; s > 0; s >>= 1) {
        if (tid < s) red[tid] += red[tid + s];
        __syncthreads();
    }
    if (tid == 0) {
        float y2 = red[0];
        float sigma = y2 / (sqrtf(y2) + EPS_SN);
        inv_sigma[0] = 1.f / sigma;
    }
}

// ---------------------------------------------------------------------------
// wbf = bf16(w * inv_sigma)
// ---------------------------------------------------------------------------
__global__ void scale_w_bf16(const float* __restrict__ w,
                             const float* __restrict__ inv_sigma,
                             __bf16* __restrict__ out, int total) {
    int i = blockIdx.x * 256 + threadIdx.x;
    if (i < total) out[i] = (__bf16)(w[i] * inv_sigma[0]);
}

// ---------------------------------------------------------------------------
// Projection GEMM on WMMA: out = (w/sigma)[M x C] @ x[b][C][N] + bias
//   MTILES = M/16 (2 for Q/K, 16 for V); block covers (b, 64-col n-slab).
//   x slab staged transposed in LDS as [n][cc] (row stride 40 elems = 80 B,
//   16B-aligned rows for B-fragment vector loads).
//   TOUT=false: out[b][m][n]  (V layout);  TOUT=true: out[b][n][m] (Q/K).
// ---------------------------------------------------------------------------
template <int MTILES, bool TOUT>
__global__ __launch_bounds__(256)
void proj_gemm_kernel(const float* __restrict__ x,
                      const __bf16* __restrict__ wbf,
                      const float* __restrict__ bias,
                      __bf16* __restrict__ outp,
                      int C, int N) {
    constexpr int TILES = MTILES * 4;           // 16x16 D tiles per block
    constexpr int TPW   = (TILES + 7) / 8;      // tiles per wave (1 or 8)
    __shared__ __bf16 xt[64 * 40];              // [n][cc], padded

    const int tid  = threadIdx.x;
    const int lane = tid & 31;
    const int wave = tid >> 5;
    const int ll = lane & 15;
    const int lh = lane >> 4;

    const int nblocks = N / 64;
    const int b  = blockIdx.x / nblocks;
    const int n0 = (blockIdx.x % nblocks) * 64;

    const float* xb = x + ((long)b * C) * N;
    const int M = MTILES * 16;

    v8f acc[TPW];
#pragma unroll
    for (int q = 0; q < TPW; ++q)
#pragma unroll
        for (int r = 0; r < 8; ++r) acc[q][r] = 0.f;

    for (int k0 = 0; k0 < 256; k0 += 32) {
        __syncthreads();
        // stage x^T tile: 32 cc x 64 n -> xt[n][cc]  (coalesced reads over n)
#pragma unroll
        for (int rep = 0; rep < 8; ++rep) {
            const int idx = rep * 256 + tid;
            const int cc = idx >> 6;
            const int n  = idx & 63;
            xt[n * 40 + cc] = (__bf16)xb[(long)(k0 + cc) * N + n0 + n];
        }
        __syncthreads();

#pragma unroll
        for (int q = 0; q < TPW; ++q) {
            const int tau = wave * TPW + q;
            const int ct = tau >> 2;
            const int snt = tau & 3;
            // A fragment: rows m = ct*16+ll, K chunks cc = k0+8*lh and +16
            const __bf16* ar = wbf + (long)(ct * 16 + ll) * 256 + k0 + 8 * lh;
            v8bf alo = *(const v8bf*)(ar);
            v8bf ahi = *(const v8bf*)(ar + 16);
            v16bf av;
#pragma unroll
            for (int e = 0; e < 8; ++e) { av[e] = alo[e]; av[8 + e] = ahi[e]; }
            // B fragment: col n = snt*16+ll, K chunks cc = 8*lh and +16
            const __bf16* br = &xt[(snt * 16 + ll) * 40 + 8 * lh];
            v8bf blo = *(const v8bf*)(br);
            v8bf bhi = *(const v8bf*)(br + 16);
            v16bf bx;
#pragma unroll
            for (int e = 0; e < 8; ++e) { bx[e] = blo[e]; bx[8 + e] = bhi[e]; }
            acc[q] = __builtin_amdgcn_wmma_f32_16x16x32_bf16(
                false, av, false, bx, (short)0, acc[q], false, false);
        }
    }

    // epilogue: + bias, store bf16
#pragma unroll
    for (int q = 0; q < TPW; ++q) {
        const int tau = wave * TPW + q;
        const int ct = tau >> 2;
        const int snt = tau & 3;
        const int nn = n0 + snt * 16 + ll;
#pragma unroll
        for (int r = 0; r < 8; ++r) {
            const int c = ct * 16 + r + 8 * lh;
            const float val = acc[q][r] + bias[c];
            if (TOUT) {
                outp[((long)b * N + nn) * (long)M + c] = (__bf16)val;  // [n][m]
            } else {
                outp[((long)b * 256 + c) * N + nn] = (__bf16)val;      // [m][n]
            }
        }
    }
}

// ---------------------------------------------------------------------------
// Issue one j-tile's cooperative async global->LDS loads.
// Uniform 5 async ops per thread: 1x b64 (K tile 32x32 bf16 = 2KB),
// 4x b128 (V tile 256x32 bf16 = 16KB, one 64B row per thread).
// ---------------------------------------------------------------------------
__device__ __forceinline__ void issue_tile_async(const __bf16* __restrict__ kg,
                                                 const __bf16* __restrict__ vg,
                                                 int j0, int N, int tid,
                                                 unsigned kbuf_off,
                                                 unsigned vbuf_off) {
    {
        const char* g = (const char*)kg + ((long)(j0 + (tid >> 3)) * 32) * 2 + (tid & 7) * 8;
        unsigned l = kbuf_off + (unsigned)((tid >> 3) * 64 + (tid & 7) * 8);
        asm volatile("global_load_async_to_lds_b64 %0, %1, off"
                     :: "v"(l), "v"(g) : "memory");
    }
    {
        const char* g = (const char*)vg + ((long)tid * N + j0) * 2;
        unsigned l = vbuf_off + (unsigned)(tid * 64);
#pragma unroll
        for (int c = 0; c < 4; ++c) {
            asm volatile("global_load_async_to_lds_b128 %0, %1, off"
                         :: "v"(l + c * 16), "v"(g + c * 16) : "memory");
        }
    }
}

// ---------------------------------------------------------------------------
// Flash attention, transposed WMMA formulation:
//   S' = K_tile x Q      -> D[M=j, N=i]  (lane owns query column i)
//   O' = V_tile x P      -> D[M=ch, N=i] (softmax stats are per-lane scalars)
// ---------------------------------------------------------------------------
__global__ __launch_bounds__(256)
void attn_kernel(const __bf16* __restrict__ Q,
                 const __bf16* __restrict__ K,
                 const __bf16* __restrict__ V,
                 const float* __restrict__ x,
                 const float* __restrict__ gamma,
                 float* __restrict__ out,
                 int C, int N) {
    __shared__ __bf16 kbuf[2][32 * 32];    //  4 KB: [j][d]
    __shared__ __bf16 vbuf[2][256 * 32];   // 32 KB: [ch][j]
    __shared__ __bf16 pbuf[8][16 * 32];    //  8 KB: per-wave P, [i][j]

    const int tid  = threadIdx.x;
    const int lane = tid & 31;
    const int wave = tid >> 5;
    const int lh = lane >> 4;
    const int ll = lane & 15;

    const int blocksPerBatch = N / 128;
    const int b  = blockIdx.x / blocksPerBatch;
    const int i0 = (blockIdx.x % blocksPerBatch) * 128 + wave * 16;

    const __bf16* kbase = K + ((long)b * N) * 32;
    const __bf16* vbase = V + ((long)b * C) * N;

    const unsigned koff[2] = { (unsigned)(uintptr_t)&kbuf[0][0],
                               (unsigned)(uintptr_t)&kbuf[1][0] };
    const unsigned voff[2] = { (unsigned)(uintptr_t)&vbuf[0][0],
                               (unsigned)(uintptr_t)&vbuf[1][0] };

    // Q as B-matrix fragment (constant): lane col i = i0+ll, d chunks 8*lh and +16
    v16bf bq;
    {
        const __bf16* qp = Q + ((long)(b * N + i0 + ll)) * 32 + 8 * lh;
        v8bf lo = *(const v8bf*)(qp);
        v8bf hi = *(const v8bf*)(qp + 16);
#pragma unroll
        for (int e = 0; e < 8; ++e) { bq[e] = lo[e]; bq[8 + e] = hi[e]; }
    }

    v8f oacc[16];
#pragma unroll
    for (int t = 0; t < 16; ++t)
#pragma unroll
        for (int r = 0; r < 8; ++r) oacc[t][r] = 0.f;

    float m = -1e30f, lsum = 0.f;
    __bf16* pw = &pbuf[wave][0];

    const int nsteps = N / 32;
    issue_tile_async(kbase, vbase, 0, N, tid, koff[0], voff[0]);

    for (int k = 0; k < nsteps; ++k) {
        const int buf = k & 1;

        if (k + 1 < nsteps) {
            issue_tile_async(kbase, vbase, (k + 1) * 32, N, tid, koff[buf ^ 1], voff[buf ^ 1]);
            asm volatile("s_wait_asynccnt 0x5" ::: "memory");
        } else {
            asm volatile("s_wait_asynccnt 0x0" ::: "memory");
        }
        __syncthreads();

        const __bf16* kb = &kbuf[buf][0];
        const __bf16* vb = &vbuf[buf][0];

        // S' = K_tile x Q : two WMMAs, D[M=j, N=i]
        v8f s[2];
#pragma unroll
        for (int jt = 0; jt < 2; ++jt) {
            const __bf16* kr = kb + (jt * 16 + ll) * 32 + 8 * lh;
            v8bf alo = *(const v8bf*)(kr);
            v8bf ahi = *(const v8bf*)(kr + 16);
            v16bf ak;
#pragma unroll
            for (int e = 0; e < 8; ++e) { ak[e] = alo[e]; ak[8 + e] = ahi[e]; }
            v8f cz;
#pragma unroll
            for (int e = 0; e < 8; ++e) cz[e] = 0.f;
            s[jt] = __builtin_amdgcn_wmma_f32_16x16x32_bf16(
                false, ak, false, bq, (short)0, cz, false, false);
        }

        // online softmax: per-lane regs + one half-exchange
        float mx = s[0][0];
#pragma unroll
        for (int r = 1; r < 8; ++r) mx = fmaxf(mx, s[0][r]);
#pragma unroll
        for (int r = 0; r < 8; ++r) mx = fmaxf(mx, s[1][r]);
        mx = fmaxf(mx, __shfl_xor(mx, 16));
        const float mn = fmaxf(m, mx);
        const float scale = __expf(m - mn);
        m = mn;
        float rs = 0.f;
#pragma unroll
        for (int jt = 0; jt < 2; ++jt)
#pragma unroll
            for (int r = 0; r < 8; ++r) {
                const float p = __expf(s[jt][r] - mn);
                s[jt][r] = p;
                rs += p;
            }
        rs += __shfl_xor(rs, 16);
        lsum = lsum * scale + rs;

        // P -> LDS [i][j] via contiguous 16B stores
#pragma unroll
        for (int jt = 0; jt < 2; ++jt) {
            v8bf pk;
#pragma unroll
            for (int r = 0; r < 8; ++r) pk[r] = (__bf16)s[jt][r];
            *(v8bf*)(pw + ll * 32 + 8 * lh + 16 * jt) = pk;
        }

        // rescale O (per-lane scalar)
#pragma unroll
        for (int t = 0; t < 16; ++t)
#pragma unroll
            for (int r = 0; r < 8; ++r) oacc[t][r] *= scale;

        // P as B fragment
        v16bf bp;
        {
            const __bf16* pr = pw + ll * 32 + 8 * lh;
            v8bf plo = *(const v8bf*)(pr);
            v8bf phi = *(const v8bf*)(pr + 16);
#pragma unroll
            for (int e = 0; e < 8; ++e) { bp[e] = plo[e]; bp[8 + e] = phi[e]; }
        }

        // O' += V_tile x P over 16 channel tiles
#pragma unroll
        for (int t = 0; t < 16; ++t) {
            const __bf16* vr = vb + (t * 16 + ll) * 32 + 8 * lh;
            v8bf alo = *(const v8bf*)(vr);
            v8bf ahi = *(const v8bf*)(vr + 16);
            v16bf av;
#pragma unroll
            for (int e = 0; e < 8; ++e) { av[e] = alo[e]; av[8 + e] = ahi[e]; }
            oacc[t] = __builtin_amdgcn_wmma_f32_16x16x32_bf16(
                false, av, false, bp, (short)0, oacc[t], false, false);
        }

        __syncthreads();
    }

    // epilogue: out = gamma * (O / l) + x ; coalesced over i
    const float g = gamma[0];
    const float invl = 1.f / lsum;
    const int i = i0 + ll;
#pragma unroll
    for (int t = 0; t < 16; ++t) {
#pragma unroll
        for (int r = 0; r < 8; ++r) {
            const int ch = t * 16 + r + 8 * lh;
            const long idx = ((long)b * C + ch) * N + i;
            out[idx] = g * (oacc[t][r] * invl) + x[idx];
        }
    }
}

// ---------------------------------------------------------------------------
extern "C" void kernel_launch(void* const* d_in, const int* in_sizes, int n_in,
                              void* d_out, int out_size, void* d_ws, size_t ws_size,
                              hipStream_t stream) {
    const float* x     = (const float*)d_in[0];
    const float* wq    = (const float*)d_in[1];
    const float* bq    = (const float*)d_in[2];
    const float* uq    = (const float*)d_in[3];
    const float* wk    = (const float*)d_in[5];
    const float* bk    = (const float*)d_in[6];
    const float* uk    = (const float*)d_in[7];
    const float* wv    = (const float*)d_in[9];
    const float* bv    = (const float*)d_in[10];
    const float* uv    = (const float*)d_in[11];
    const float* gamma = (const float*)d_in[13];
    float* out = (float*)d_out;

    const int B = 8, C = 256, CQ = 32, N = 4096;

    char* ws = (char*)d_ws;
    __bf16* Qbf  = (__bf16*)ws;                          // B*N*CQ  (2 MB)
    __bf16* Kbf  = Qbf + (size_t)B * N * CQ;             // B*N*CQ  (2 MB)
    __bf16* Vbf  = Kbf + (size_t)B * N * CQ;             // B*C*N   (16 MB)
    __bf16* wqbf = Vbf + (size_t)B * C * N;              // 32*256
    __bf16* wkbf = wqbf + (size_t)CQ * C;                // 32*256
    __bf16* wvbf = wkbf + (size_t)CQ * C;                // 256*256
    float*  sig  = (float*)(wvbf + (size_t)C * C);       // 3 floats (1/sigma)

    sn_sigma_kernel<<<1, 256, 0, stream>>>(wq, uq, CQ, C, sig + 0);
    sn_sigma_kernel<<<1, 256, 0, stream>>>(wk, uk, CQ, C, sig + 1);
    sn_sigma_kernel<<<1, 256, 0, stream>>>(wv, uv, C,  C, sig + 2);

    scale_w_bf16<<<(CQ * C + 255) / 256, 256, 0, stream>>>(wq, sig + 0, wqbf, CQ * C);
    scale_w_bf16<<<(CQ * C + 255) / 256, 256, 0, stream>>>(wk, sig + 1, wkbf, CQ * C);
    scale_w_bf16<<<(C * C + 255) / 256, 256, 0, stream>>>(wv, sig + 2, wvbf, C * C);

    // WMMA projection GEMMs (Q/K: [n][d] layout; V: [c][n] layout)
    proj_gemm_kernel<2,  true ><<<B * (N / 64), 256, 0, stream>>>(x, wqbf, bq, Qbf, C, N);
    proj_gemm_kernel<2,  true ><<<B * (N / 64), 256, 0, stream>>>(x, wkbf, bk, Kbf, C, N);
    proj_gemm_kernel<16, false><<<B * (N / 64), 256, 0, stream>>>(x, wvbf, bv, Vbf, C, N);

    attn_kernel<<<B * N / 128, 256, 0, stream>>>(Qbf, Kbf, Vbf, x, gamma, out, C, N);
}
